// SpatialSelfAttention_11407433138216
// MI455X (gfx1250) — matrix operand order repaired and checked by reading, not compiled, not run on hardware
//
#include <hip/hip_runtime.h>

// ---------------------------------------------------------------------------
// Spatial self-attention block for MI455X (gfx1250, wave32, WMMA).
// B=2, C=512, H=W=64 -> N=4096 tokens, 8 heads x 64 dim.
// All matmuls run through v_wmma_f32_16x16x32_f16 (f16 in, f32 accum).
// Flash-style attention avoids the 1GB score tensor; all intermediates live
// in d_ws (~42.5MB) and stay resident in the 192MB L2.
// K/V tiles are staged once per block into LDS via gfx1250 async global->LDS
// copies (ASYNCcnt), double-buffered against WMMA compute.
// ---------------------------------------------------------------------------

#define BB 2
#define CC 512
#define NN 4096          // H*W
#define NHEADS 8
#define HD 64            // head dim
#define NTOK (BB * NN)   // 8192 tokens
#define C3 (3 * CC)      // 1536
#define EPSV 1e-5f
#define QSCALE 0.125f    // HD^-0.5

typedef __attribute__((ext_vector_type(16))) _Float16 v16h;
typedef __attribute__((ext_vector_type(8)))  float    v8f;
typedef int v4i_vs __attribute__((vector_size(16)));   // payload type for b128 async copy
using half_t = _Float16;

#if __has_builtin(__builtin_amdgcn_global_load_async_to_lds_b128)
#define ASYNC_LDS 1
#else
#define ASYNC_LDS 0
#endif

__device__ __forceinline__ v8f wmma_f16(v16h a, v16h b, v8f c) {
  // D = A(16x32 f16) * B(32x16 f16) + C(16x16 f32)
  return __builtin_amdgcn_wmma_f32_16x16x32_f16(
      /*neg_a=*/false, a, /*neg_b=*/false, b,
      /*c_mod=*/(short)0, c, /*reuse_a=*/false, /*reuse_b=*/false);
}

// 32 bytes -> one f16 fragment register block (2x b128 loads, global or LDS)
__device__ __forceinline__ v16h load_frag(const half_t* p0, const half_t* p1) {
  union { v16h v; uint4 u[2]; } t;
  t.u[0] = *(const uint4*)p0;
  t.u[1] = *(const uint4*)p1;
  return t.v;
}

// One 16-byte global -> LDS copy per lane (async on gfx1250).
__device__ __forceinline__ void copy16B_to_lds(const half_t* g, half_t* l) {
#if ASYNC_LDS
  __builtin_amdgcn_global_load_async_to_lds_b128(
      (__attribute__((address_space(1))) v4i_vs*)g,
      (__attribute__((address_space(3))) v4i_vs*)l, 0, 0);
#else
  *(uint4*)l = *(const uint4*)g;   // sync fallback: global->VGPR->ds_store
#endif
}

__device__ __forceinline__ void wait_stage() {
#if ASYNC_LDS
  asm volatile("s_wait_asynccnt 0x0" ::: "memory");
#endif
  // sync fallback: __syncthreads() below already waits dscnt/loadcnt
}

// ---------------------------------------------------------------------------
// Kernel 0: x [B,C,N] f32 -> tok16 [B*N, C] f16 (tiled LDS transpose)
// ---------------------------------------------------------------------------
__global__ __launch_bounds__(256) void k_pack_tokens(const float* __restrict__ x,
                                                     half_t* __restrict__ tok16) {
  __shared__ float tile[32][33];
  const int b  = blockIdx.z;
  const int c0 = blockIdx.y * 32;
  const int n0 = blockIdx.x * 32;
  const int tx = threadIdx.x & 31;
  const int ty = threadIdx.x >> 5;   // 0..7
  const float* xb = x + (size_t)b * CC * NN;
#pragma unroll
  for (int i = 0; i < 4; ++i) {
    const int c = c0 + ty + i * 8;
    tile[ty + i * 8][tx] = xb[(size_t)c * NN + (n0 + tx)];
  }
  __syncthreads();
  half_t* tb = tok16 + (size_t)b * NN * CC;
#pragma unroll
  for (int i = 0; i < 4; ++i) {
    const int n = n0 + ty + i * 8;
    tb[(size_t)n * CC + (c0 + tx)] = (half_t)tile[tx][ty + i * 8];
  }
}

// ---------------------------------------------------------------------------
// Kernel 1: weight f32 -> f16
// ---------------------------------------------------------------------------
__global__ __launch_bounds__(256) void k_convert_w(const float* __restrict__ qkv_w,
                                                   const float* __restrict__ out_w,
                                                   half_t* __restrict__ qkv_w16,
                                                   half_t* __restrict__ out_w16) {
  const int i = blockIdx.x * blockDim.x + threadIdx.x;
  if (i < C3 * CC) qkv_w16[i] = (half_t)qkv_w[i];
  if (i < CC * CC) out_w16[i] = (half_t)out_w[i];
}

// ---------------------------------------------------------------------------
// Kernel 2: fused QKV GEMM.  [8192,512] @ qkv_w^T -> scatter into
//   q16 [BH,N,D] (pre-scaled), k16 [BH,N,D], vT16 [BH,D,N].
// Wave computes 16 rows x 64 cols; 4 waves/block cover 256 cols.
// ---------------------------------------------------------------------------
__global__ __launch_bounds__(128) void k_qkv_gemm(const half_t* __restrict__ tok16,
                                                  const half_t* __restrict__ w16,
                                                  const float* __restrict__ bias,
                                                  half_t* __restrict__ q16,
                                                  half_t* __restrict__ k16,
                                                  half_t* __restrict__ vT16) {
  const int wave  = threadIdx.x >> 5;
  const int lane  = threadIdx.x & 31;
  const int lhalf = lane >> 4;
  const int l16   = lane & 15;

  const int row0 = blockIdx.x * 16;                 // token rows
  const int col0 = blockIdx.y * 256 + wave * 64;    // output cols in [0,1536)

  v8f acc[4] = {};
  const half_t* arow = tok16 + (size_t)(row0 + l16) * CC;

  for (int t = 0; t < CC; t += 32) {
    const half_t* ap = arow + t + lhalf * 8;
    const v16h a = load_frag(ap, ap + 16);
#pragma unroll
    for (int jc = 0; jc < 4; ++jc) {
      const half_t* bp = w16 + (size_t)(col0 + jc * 16 + l16) * CC + t + lhalf * 8;
      acc[jc] = wmma_f16(a, load_frag(bp, bp + 16), acc[jc]);
    }
  }

  const int b  = row0 / NN;
  const int n0 = row0 % NN;
#pragma unroll
  for (int jc = 0; jc < 4; ++jc) {
    const int col    = col0 + jc * 16 + l16;
    const int sel    = col / CC;           // 0=q 1=k 2=v
    const int within = col % CC;
    const int h      = within / HD;
    const int d      = within % HD;
    const float bv   = bias[col];
    const size_t bh  = (size_t)(b * NHEADS + h);
#pragma unroll
    for (int r = 0; r < 8; ++r) {
      const int n   = n0 + r + 8 * lhalf;
      const float v = acc[jc][r] + bv;
      if (sel == 0)      q16[(bh * NN + n) * HD + d]  = (half_t)(v * QSCALE);
      else if (sel == 1) k16[(bh * NN + n) * HD + d]  = (half_t)v;
      else               vT16[(bh * HD + d) * NN + n] = (half_t)v;
    }
  }
}

// ---------------------------------------------------------------------------
// Kernel 3: flash attention.  Block = 4 waves = 64 query rows; K/V tiles of
// 32 keys staged once per block into double-buffered LDS (async copy overlaps
// compute).  QK^T and P@V via WMMA; online softmax with shfl_xor row
// reductions; P re-laid C-frag -> A-frag through per-wave LDS.
// ---------------------------------------------------------------------------
__global__ __launch_bounds__(128) void k_attn(const half_t* __restrict__ q16,
                                              const half_t* __restrict__ k16,
                                              const half_t* __restrict__ vT16,
                                              half_t* __restrict__ attn16) {
  __shared__ __align__(64) half_t kbuf[2][32][64];   // 2 x 4KB: keys x headdim
  __shared__ __align__(64) half_t vtbuf[2][64][32];  // 2 x 4KB: headdim x keys
  __shared__ __align__(64) half_t pbuf[4][16][32];   // per-wave P transpose
  const int tid   = threadIdx.x;
  const int wave  = tid >> 5;
  const int lane  = tid & 31;
  const int lhalf = lane >> 4;
  const int l16   = lane & 15;

  const int bh = blockIdx.y;                       // b*NHEADS + h
  const int q0 = blockIdx.x * 64 + wave * 16;      // query tile base (within batch)

  const half_t* Q  = q16  + (size_t)bh * NN * HD;
  const half_t* K  = k16  + (size_t)bh * NN * HD;
  const half_t* Vt = vT16 + (size_t)bh * HD * NN;

  // cooperative stage of one 32-key K/V tile into LDS (512 x 16B chunks)
  auto stage = [&](int key0, int buf) {
    const half_t* ksrc = K + (size_t)key0 * HD;    // contiguous 4KB
#pragma unroll
    for (int c = 0; c < 2; ++c) {
      const int idx = tid + c * 128;               // 16B chunk id 0..255
      copy16B_to_lds(ksrc + idx * 8, &kbuf[buf][0][0] + idx * 8);
    }
#pragma unroll
    for (int c = 0; c < 2; ++c) {
      const int idx = tid + c * 128;
      const int d = idx >> 2, j = idx & 3;         // row d (64B), 16B chunk j
      copy16B_to_lds(Vt + (size_t)d * NN + key0 + j * 8, &vtbuf[buf][d][j * 8]);
    }
  };

  // Resident Q fragments: 16 rows x 64 dim = 2 A-frags
  v16h qa[2];
#pragma unroll
  for (int t = 0; t < 2; ++t) {
    const half_t* p = Q + (size_t)(q0 + l16) * HD + t * 32 + lhalf * 8;
    qa[t] = load_frag(p, p + 16);
  }

  float m[8], lsum[8];
  v8f o[4] = {};
#pragma unroll
  for (int r = 0; r < 8; ++r) { m[r] = -1e30f; lsum[r] = 0.f; }

  stage(0, 0);
  wait_stage();
  __syncthreads();

  for (int key0 = 0; key0 < NN; key0 += 32) {
    const int cur = (key0 >> 5) & 1;
    if (key0 + 32 < NN) stage(key0 + 32, cur ^ 1);   // prefetch next tile

    // S = Q K^T  (16 x 32 as two 16x16 frags) from LDS
    v8f s[2] = {};
#pragma unroll
    for (int ks = 0; ks < 2; ++ks) {
#pragma unroll
      for (int t = 0; t < 2; ++t) {
        const half_t* p = &kbuf[cur][ks * 16 + l16][t * 32 + lhalf * 8];
        s[ks] = wmma_f16(qa[t], load_frag(p, p + 16), s[ks]);
      }
    }
    // online softmax (rows live per-VGPR; columns across 16-lane halves)
    float alpha[8];
#pragma unroll
    for (int r = 0; r < 8; ++r) {
      float mx = fmaxf(s[0][r], s[1][r]);
      mx = fmaxf(mx, __shfl_xor(mx, 1, 32));
      mx = fmaxf(mx, __shfl_xor(mx, 2, 32));
      mx = fmaxf(mx, __shfl_xor(mx, 4, 32));
      mx = fmaxf(mx, __shfl_xor(mx, 8, 32));
      const float mn = fmaxf(m[r], mx);
      alpha[r] = __expf(m[r] - mn);
      m[r] = mn;
      const float p0 = __expf(s[0][r] - mn);
      const float p1 = __expf(s[1][r] - mn);
      s[0][r] = p0; s[1][r] = p1;
      float rs = p0 + p1;
      rs += __shfl_xor(rs, 1, 32);
      rs += __shfl_xor(rs, 2, 32);
      rs += __shfl_xor(rs, 4, 32);
      rs += __shfl_xor(rs, 8, 32);
      lsum[r] = lsum[r] * alpha[r] + rs;
    }
#pragma unroll
    for (int dc = 0; dc < 4; ++dc)
#pragma unroll
      for (int r = 0; r < 8; ++r) o[dc][r] *= alpha[r];

    // P: C-frag layout -> LDS (row-major 16x32 f16) -> A-frag layout
#pragma unroll
    for (int ks = 0; ks < 2; ++ks)
#pragma unroll
      for (int r = 0; r < 8; ++r)
        pbuf[wave][r + 8 * lhalf][ks * 16 + l16] = (half_t)s[ks][r];
    asm volatile("s_wait_dscnt 0x0" ::: "memory");   // per-wave LDS RAW
    const half_t* pp = &pbuf[wave][l16][lhalf * 8];
    const v16h pa = load_frag(pp, pp + 16);

    // O += P @ V  from LDS (vT layout => two 16B runs per lane)
#pragma unroll
    for (int dc = 0; dc < 4; ++dc) {
      const half_t* p = &vtbuf[cur][dc * 16 + l16][lhalf * 8];
      o[dc] = wmma_f16(pa, load_frag(p, p + 16), o[dc]);
    }

    wait_stage();      // next tile's async copies complete
    __syncthreads();   // all waves done with 'cur' before it is reused
  }

  // epilogue: normalize and store attn16 [B*N, C] f16
  const int b = bh / NHEADS;
  const int h = bh % NHEADS;
#pragma unroll
  for (int r = 0; r < 8; ++r) {
    const float inv = 1.0f / lsum[r];
    const int n = q0 + r + 8 * lhalf;
#pragma unroll
    for (int dc = 0; dc < 4; ++dc) {
      const int c = h * HD + dc * 16 + l16;
      attn16[((size_t)(b * NN + n)) * CC + c] = (half_t)(o[dc][r] * inv);
    }
  }
}

// ---------------------------------------------------------------------------
// Kernel 4: out-proj GEMM + bias + residual(x) + LayerNorm + transposed store.
// Block = 16 tokens x 512 channels, 8 waves (each 16x64 WMMA tile).
// ---------------------------------------------------------------------------
__global__ __launch_bounds__(256) void k_outproj_ln(const half_t* __restrict__ attn16,
                                                    const half_t* __restrict__ w16,
                                                    const float* __restrict__ out_b,
                                                    const float* __restrict__ ln_g,
                                                    const float* __restrict__ ln_b,
                                                    const float* __restrict__ x,
                                                    float* __restrict__ y) {
  __shared__ float res[16][CC + 4];
  __shared__ float stats[16][2];
  const int wave  = threadIdx.x >> 5;
  const int lane  = threadIdx.x & 31;
  const int lhalf = lane >> 4;
  const int l16   = lane & 15;
  const int row0  = blockIdx.x * 16;     // global token rows
  const int b     = row0 / NN;
  const int n0    = row0 % NN;
  const int col0  = wave * 64;

  v8f acc[4] = {};
  const half_t* arow = attn16 + (size_t)(row0 + l16) * CC;
  for (int t = 0; t < CC; t += 32) {
    const half_t* ap = arow + t + lhalf * 8;
    const v16h a = load_frag(ap, ap + 16);
#pragma unroll
    for (int jc = 0; jc < 4; ++jc) {
      const half_t* bp = w16 + (size_t)(col0 + jc * 16 + l16) * CC + t + lhalf * 8;
      acc[jc] = wmma_f16(a, load_frag(bp, bp + 16), acc[jc]);
    }
  }

  const float* xb = x + (size_t)b * CC * NN;
#pragma unroll
  for (int jc = 0; jc < 4; ++jc) {
    const int c    = col0 + jc * 16 + l16;
    const float bv = out_b[c];
#pragma unroll
    for (int r = 0; r < 8; ++r) {
      const int rr = r + 8 * lhalf;
      res[rr][c] = acc[jc][r] + bv + xb[(size_t)c * NN + (n0 + rr)];
    }
  }
  __syncthreads();

  {  // LayerNorm stats: 16 threads per token row, shfl_xor reduce in halves
    const int row = threadIdx.x >> 4;
    const int seg = threadIdx.x & 15;
    float s = 0.f, sq = 0.f;
#pragma unroll
    for (int i = 0; i < 32; ++i) {
      const float v = res[row][seg * 32 + i];
      s += v; sq += v * v;
    }
    s += __shfl_xor(s, 1, 32); sq += __shfl_xor(sq, 1, 32);
    s += __shfl_xor(s, 2, 32); sq += __shfl_xor(sq, 2, 32);
    s += __shfl_xor(s, 4, 32); sq += __shfl_xor(sq, 4, 32);
    s += __shfl_xor(s, 8, 32); sq += __shfl_xor(sq, 8, 32);
    if (seg == 0) {
      const float mean = s * (1.0f / CC);
      const float var  = sq * (1.0f / CC) - mean * mean;
      stats[row][0] = mean;
      stats[row][1] = rsqrtf(var + EPSV);
    }
  }
  __syncthreads();

  {  // transposed store: consecutive threads -> consecutive tokens (coalesced)
    const int token = threadIdx.x & 15;
    const int cbase = threadIdx.x >> 4;
    const float mean = stats[token][0];
    const float rstd = stats[token][1];
    float* yb = y + (size_t)b * CC * NN + n0 + token;
#pragma unroll
    for (int i = 0; i < 32; ++i) {
      const int c = cbase + 16 * i;
      yb[(size_t)c * NN] = (res[token][c] - mean) * rstd * ln_g[c] + ln_b[c];
    }
  }
}

// ---------------------------------------------------------------------------
extern "C" void kernel_launch(void* const* d_in, const int* in_sizes, int n_in,
                              void* d_out, int out_size, void* d_ws, size_t ws_size,
                              hipStream_t stream) {
  const float* x     = (const float*)d_in[0];
  const float* qkv_w = (const float*)d_in[1];
  const float* qkv_b = (const float*)d_in[2];
  const float* out_w = (const float*)d_in[3];
  const float* out_b = (const float*)d_in[4];
  const float* ln_g  = (const float*)d_in[5];
  const float* ln_b  = (const float*)d_in[6];
  float* y = (float*)d_out;

  // workspace carve (all 16B aligned; total ~42.5 MB, L2-resident)
  char* ws = (char*)d_ws;
  half_t* tok16  = (half_t*)ws; ws += (size_t)NTOK * CC * sizeof(half_t);
  half_t* qkvw16 = (half_t*)ws; ws += (size_t)C3 * CC * sizeof(half_t);
  half_t* outw16 = (half_t*)ws; ws += (size_t)CC * CC * sizeof(half_t);
  half_t* q16    = (half_t*)ws; ws += (size_t)NTOK * CC * sizeof(half_t);
  half_t* k16    = (half_t*)ws; ws += (size_t)NTOK * CC * sizeof(half_t);
  half_t* vT16   = (half_t*)ws; ws += (size_t)NTOK * CC * sizeof(half_t);
  half_t* attn16 = (half_t*)ws; ws += (size_t)NTOK * CC * sizeof(half_t);

  k_pack_tokens<<<dim3(NN / 32, CC / 32, BB), 256, 0, stream>>>(x, tok16);
  k_convert_w<<<(C3 * CC + 255) / 256, 256, 0, stream>>>(qkv_w, out_w, qkvw16, outw16);
  k_qkv_gemm<<<dim3(NTOK / 16, C3 / 256), 128, 0, stream>>>(tok16, qkvw16, qkv_b,
                                                            q16, k16, vT16);
  k_attn<<<dim3(NN / 64, BB * NHEADS), 128, 0, stream>>>(q16, k16, vT16, attn16);
  k_outproj_ln<<<NTOK / 16, 256, 0, stream>>>(attn16, outw16, out_b, ln_g, ln_b, x, y);
}
